// Attention_84258668413036
// MI455X (gfx1250) — matrix operand order repaired and checked by reading, not compiled, not run on hardware
//
#include <hip/hip_runtime.h>

// ---------------- problem constants ----------------
#define BATCH   4
#define SEQ     2048
#define DIM     768
#define HEADS   12
#define DHEAD   64
#define MTOT    (BATCH * SEQ)      // 8192 rows of the big GEMMs

typedef __attribute__((ext_vector_type(16))) __bf16 bf16x16;
typedef __attribute__((ext_vector_type(8)))  float  fx8;

__device__ __forceinline__ unsigned short f32_to_bf16(float f) {
  union { float f; unsigned u; } v; v.f = f;
  unsigned u = v.u;
  u += 0x7FFFu + ((u >> 16) & 1u);      // round-to-nearest-even
  return (unsigned short)(u >> 16);
}

union Frag16 {
  bf16x16 v;
  uint4   q[2];
};

// A-matrix fragment (16x32 bf16) from a row-major source.
// lane holds row = lane&15; half = lane>>4.
// elems 0..7  <-> k = k0 + 8*half + {0..7}
// elems 8..15 <-> k = k0 + 16 + 8*half + {0..7}
__device__ __forceinline__ Frag16 load_fragA(const unsigned short* base, int stride,
                                             int row, int k0, int hf) {
  Frag16 f;
  const unsigned short* p = base + row * stride + k0 + 8 * hf;
  f.q[0] = *(const uint4*)(p);
  f.q[1] = *(const uint4*)(p + 16);
  return f;
}

// B-matrix fragment (32x16 bf16); source is B^T row-major (one row per output col n).
// lane holds n = lane&15; elems 0..7 <-> k = k0 + 8*half + {0..7}, elems 8..15 <-> k = k0 + 16 + 8*half + {0..7}
__device__ __forceinline__ Frag16 load_fragB(const unsigned short* base, int stride,
                                             int nrow, int k0, int hf) {
  Frag16 f;
  const unsigned short* p = base + nrow * stride + k0 + 8 * hf;
  f.q[0] = *(const uint4*)(p);
  f.q[1] = *(const uint4*)(p + 16);
  return f;
}

__device__ __forceinline__ fx8 wmma_bf16(const Frag16& a, const Frag16& b, fx8 c) {
  return __builtin_amdgcn_wmma_f32_16x16x32_bf16(false, a.v, false, b.v,
                                                 (short)0, c, false, false);
}

// Max-reduction over each 16-lane row (xor butterfly 1,2,4,8).
// Pure VALU via v_permlane16_b32 (no DS pipeline traffic in the hot loop).
__device__ __forceinline__ float rowmax16(float t) {
#if __has_builtin(__builtin_amdgcn_permlane16)
  unsigned u, p;
  u = __float_as_uint(t);
  p = __builtin_amdgcn_permlane16(u, u, 0x67452301u, 0xEFCDAB89u, false, false); // xor 1
  t = fmaxf(t, __uint_as_float(p)); u = __float_as_uint(t);
  p = __builtin_amdgcn_permlane16(u, u, 0x54761032u, 0xDCFE98BAu, false, false); // xor 2
  t = fmaxf(t, __uint_as_float(p)); u = __float_as_uint(t);
  p = __builtin_amdgcn_permlane16(u, u, 0x32107654u, 0xBA98FEDCu, false, false); // xor 4
  t = fmaxf(t, __uint_as_float(p)); u = __float_as_uint(t);
  p = __builtin_amdgcn_permlane16(u, u, 0xFEDCBA98u, 0x76543210u, false, false); // xor 8
  t = fmaxf(t, __uint_as_float(p));
#else
  t = fmaxf(t, __shfl_xor(t, 1, 32));
  t = fmaxf(t, __shfl_xor(t, 2, 32));
  t = fmaxf(t, __shfl_xor(t, 4, 32));
  t = fmaxf(t, __shfl_xor(t, 8, 32));
#endif
  return t;
}

// ---------------- conversion kernels ----------------
__global__ void cvt_x_kernel(const float* __restrict__ in,
                             unsigned short* __restrict__ out, int count) {
  int i = blockIdx.x * 256 + threadIdx.x;
  if (i < count) out[i] = f32_to_bf16(in[i]);
}

// W [K=768, N=768] f32 -> Wt [N, K] bf16 (optionally scaled)
__global__ void cvt_wT_kernel(const float* __restrict__ W,
                              unsigned short* __restrict__ out, float scale) {
  int i = blockIdx.x * 256 + threadIdx.x;
  if (i >= DIM * DIM) return;
  int n = i / DIM, k = i % DIM;
  out[i] = f32_to_bf16(W[k * DIM + n] * scale);
}

// ---------------- GEMM: C[M,768] = A[M,768] x Bt[768,768]^T ----------------
// MODE 0: store bf16 head-split [b,h,n,d]   (Q, K)
// MODE 1: store bf16 head-split transposed [b,h,d,n]   (V)
// MODE 2: store f32 row-major + bias        (final projection)
template <int MODE>
__global__ __launch_bounds__(128) void gemm_kernel(const unsigned short* __restrict__ A,
                                                   const unsigned short* __restrict__ Bt,
                                                   void* __restrict__ Out,
                                                   const float* __restrict__ bias) {
  const int lane = threadIdx.x & 31;
  const int wave = threadIdx.x >> 5;
  const int hf = lane >> 4;
  const int ln = lane & 15;
  const int m0 = blockIdx.x * 128 + (wave & 1) * 64;   // 4 m-tiles of 16
  const int n0 = blockIdx.y * 64 + (wave >> 1) * 32;   // 2 n-tiles of 16

  fx8 acc[4][2];
#pragma unroll
  for (int i = 0; i < 4; ++i)
#pragma unroll
    for (int j = 0; j < 2; ++j)
      acc[i][j] = (fx8){0.f, 0.f, 0.f, 0.f, 0.f, 0.f, 0.f, 0.f};

  for (int k0 = 0; k0 < DIM; k0 += 32) {
    Frag16 a[4], b[2];
#pragma unroll
    for (int i = 0; i < 4; ++i)
      a[i] = load_fragA(A, DIM, m0 + i * 16 + ln, k0, hf);
#pragma unroll
    for (int j = 0; j < 2; ++j)
      b[j] = load_fragB(Bt, DIM, n0 + j * 16 + ln, k0, hf);
#pragma unroll
    for (int i = 0; i < 4; ++i)
#pragma unroll
      for (int j = 0; j < 2; ++j)
        acc[i][j] = wmma_bf16(a[i], b[j], acc[i][j]);
  }

#pragma unroll
  for (int i = 0; i < 4; ++i)
#pragma unroll
    for (int j = 0; j < 2; ++j)
#pragma unroll
      for (int r = 0; r < 8; ++r) {
        int gm = m0 + i * 16 + r + 8 * hf;   // global row (b*seq + q)
        int gn = n0 + j * 16 + ln;           // global col (h*64 + d)
        float val = acc[i][j][r];
        if (MODE == 2) {
          ((float*)Out)[gm * DIM + gn] = val + bias[gn];
        } else {
          int bi = gm >> 11, qq = gm & (SEQ - 1);
          int hh = gn >> 6, dd = gn & (DHEAD - 1);
          size_t idx;
          if (MODE == 0)  // [b,h,n,d]
            idx = ((size_t)(bi * HEADS + hh) * SEQ + qq) * DHEAD + dd;
          else            // [b,h,d,n]
            idx = ((size_t)(bi * HEADS + hh) * DHEAD + dd) * SEQ + qq;
          ((unsigned short*)Out)[idx] = f32_to_bf16(val);
        }
      }
}

// ---------------- flash attention ----------------
// Q,K: bf16 [b,h,n,d]; V: bf16 [b,h,d,n]; Ob: bf16 [b,n,h*d]
__global__ __launch_bounds__(128) void attn_kernel(const unsigned short* __restrict__ Q,
                                                   const unsigned short* __restrict__ K,
                                                   const unsigned short* __restrict__ V,
                                                   unsigned short* __restrict__ Ob) {
  __shared__ unsigned short plds[4 * 16 * 32];   // per-wave 16x32 bf16 P tile

  const int lane = threadIdx.x & 31;
  const int wave = threadIdx.x >> 5;
  const int hf = lane >> 4;
  const int ln = lane & 15;
  const int bh = blockIdx.y;            // 0..47
  const int bi = bh / HEADS, hh = bh % HEADS;
  const int q0 = blockIdx.x * 64 + wave * 16;

  const unsigned short* Qh = Q + (size_t)bh * SEQ * DHEAD;
  const unsigned short* Kh = K + (size_t)bh * SEQ * DHEAD;
  const unsigned short* Vh = V + (size_t)bh * DHEAD * SEQ;   // row stride SEQ
  unsigned short* myl = plds + wave * (16 * 32);

  Frag16 aq0 = load_fragA(Qh, DHEAD, q0 + ln, 0, hf);
  Frag16 aq1 = load_fragA(Qh, DHEAD, q0 + ln, 32, hf);

  // constant all-ones B fragment (bf16 1.0 = 0x3F80) for row-sum WMMA
  Frag16 ones;
  ones.q[0] = (uint4){0x3F803F80u, 0x3F803F80u, 0x3F803F80u, 0x3F803F80u};
  ones.q[1] = ones.q[0];

  fx8 o[4];
#pragma unroll
  for (int t = 0; t < 4; ++t)
    o[t] = (fx8){0.f, 0.f, 0.f, 0.f, 0.f, 0.f, 0.f, 0.f};
  float me[8], le[8];
#pragma unroll
  for (int r = 0; r < 8; ++r) { me[r] = -1e30f; le[r] = 0.f; }

  for (int j0 = 0; j0 < SEQ; j0 += 32) {
    // S tiles: Q(16x64) . K^T -> 16x32 scores (two 16x16 tiles)
    fx8 s0 = (fx8){0.f, 0.f, 0.f, 0.f, 0.f, 0.f, 0.f, 0.f};
    fx8 s1 = s0;
    {
      Frag16 bk;
      bk = load_fragB(Kh, DHEAD, j0 + ln, 0, hf);       s0 = wmma_bf16(aq0, bk, s0);
      bk = load_fragB(Kh, DHEAD, j0 + ln, 32, hf);      s0 = wmma_bf16(aq1, bk, s0);
      bk = load_fragB(Kh, DHEAD, j0 + 16 + ln, 0, hf);  s1 = wmma_bf16(aq0, bk, s1);
      bk = load_fragB(Kh, DHEAD, j0 + 16 + ln, 32, hf); s1 = wmma_bf16(aq1, bk, s1);
    }

    float scl[8];
#pragma unroll
    for (int r = 0; r < 8; ++r) {
      float t = rowmax16(fmaxf(s0[r], s1[r]));  // per-row max within 16-lane half
      float mn = fmaxf(me[r], t);
      scl[r] = __expf(me[r] - mn);
      me[r] = mn;
      float p0 = __expf(s0[r] - mn);
      float p1 = __expf(s1[r] - mn);
      int row = r + 8 * hf;
      myl[row * 32 + ln]      = f32_to_bf16(p0);
      myl[row * 32 + 16 + ln] = f32_to_bf16(p1);
    }
#pragma unroll
    for (int t = 0; t < 4; ++t)
#pragma unroll
      for (int r = 0; r < 8; ++r)
        o[t][r] *= scl[r];

    asm volatile("s_wait_dscnt 0" ::: "memory");

    Frag16 ap = load_fragA(myl, 32, ln, 0, hf);     // P as 16x32 A-fragment

    // Row sums of P via the matrix unit: P(16x32) x ones(32x16).
    // D element r holds rowsum(row r+8h) in every column -> aligned with le[].
    fx8 rsum = (fx8){0.f, 0.f, 0.f, 0.f, 0.f, 0.f, 0.f, 0.f};
    rsum = wmma_bf16(ap, ones, rsum);
#pragma unroll
    for (int r = 0; r < 8; ++r)
      le[r] = le[r] * scl[r] + rsum[r];

#pragma unroll
    for (int t = 0; t < 4; ++t) {
      Frag16 bv = load_fragB(Vh, SEQ, t * 16 + ln, j0, hf);  // V^T rows = d cols
      o[t] = wmma_bf16(ap, bv, o[t]);
    }
  }

  float inv[8];
#pragma unroll
  for (int r = 0; r < 8; ++r) inv[r] = 1.0f / le[r];
#pragma unroll
  for (int t = 0; t < 4; ++t)
#pragma unroll
    for (int r = 0; r < 8; ++r) {
      int qq = q0 + r + 8 * hf;
      int dd = t * 16 + ln;
      Ob[((size_t)(bi * SEQ + qq)) * DIM + hh * DHEAD + dd] = f32_to_bf16(o[t][r] * inv[r]);
    }
}

// ---------------- launch ----------------
extern "C" void kernel_launch(void* const* d_in, const int* in_sizes, int n_in,
                              void* d_out, int out_size, void* d_ws, size_t ws_size,
                              hipStream_t stream) {
  const float* x  = (const float*)d_in[0];
  const float* Wq = (const float*)d_in[1];
  const float* Wk = (const float*)d_in[2];
  const float* Wv = (const float*)d_in[3];
  const float* Wo = (const float*)d_in[4];
  const float* bo = (const float*)d_in[5];

  char* ws = (char*)d_ws;
  size_t ofs = 0;
  const size_t xbytes = (size_t)MTOT * DIM * 2;       // 12.6 MB each
  const size_t wbytes = (size_t)DIM * DIM * 2;        // 1.18 MB each
  unsigned short* Xb  = (unsigned short*)(ws + ofs); ofs += xbytes;
  unsigned short* WqT = (unsigned short*)(ws + ofs); ofs += wbytes;
  unsigned short* WkT = (unsigned short*)(ws + ofs); ofs += wbytes;
  unsigned short* WvT = (unsigned short*)(ws + ofs); ofs += wbytes;
  unsigned short* WoT = (unsigned short*)(ws + ofs); ofs += wbytes;
  unsigned short* Qb  = (unsigned short*)(ws + ofs); ofs += xbytes;
  unsigned short* Kb  = (unsigned short*)(ws + ofs); ofs += xbytes;
  unsigned short* Vt  = (unsigned short*)(ws + ofs); ofs += xbytes;
  unsigned short* Ob  = (unsigned short*)(ws + ofs); ofs += xbytes;  // ~68 MB total

  const int xcount = MTOT * DIM;
  cvt_x_kernel<<<(xcount + 255) / 256, 256, 0, stream>>>(x, Xb, xcount);
  const int wcount = DIM * DIM;
  const float qscale = 0.125f;   // 1/sqrt(DHEAD) folded into Wq
  cvt_wT_kernel<<<(wcount + 255) / 256, 256, 0, stream>>>(Wq, WqT, qscale);
  cvt_wT_kernel<<<(wcount + 255) / 256, 256, 0, stream>>>(Wk, WkT, 1.0f);
  cvt_wT_kernel<<<(wcount + 255) / 256, 256, 0, stream>>>(Wv, WvT, 1.0f);
  cvt_wT_kernel<<<(wcount + 255) / 256, 256, 0, stream>>>(Wo, WoT, 1.0f);

  dim3 gg(MTOT / 128, DIM / 64);   // 64 x 12 blocks, 128 threads each
  gemm_kernel<0><<<gg, 128, 0, stream>>>(Xb, WqT, (void*)Qb, nullptr);
  gemm_kernel<0><<<gg, 128, 0, stream>>>(Xb, WkT, (void*)Kb, nullptr);
  gemm_kernel<1><<<gg, 128, 0, stream>>>(Xb, WvT, (void*)Vt, nullptr);

  attn_kernel<<<dim3(SEQ / 64, BATCH * HEADS), 128, 0, stream>>>(Qb, Kb, Vt, Ob);

  gemm_kernel<2><<<gg, 128, 0, stream>>>(Ob, WoT, d_out, bo);
}
